// SimpleTransformerCUDA_48189533061257
// MI455X (gfx1250) — matrix operand order repaired and checked by reading, not compiled, not run on hardware
//
#include <hip/hip_runtime.h>
#include <cstdint>

#define D_MODEL 1024
#define SEQ     2048
#define BATCH   2
#define HEADS   16
#define HD      64
#define MTOT    (BATCH * SEQ)   // 4096 token rows

typedef __attribute__((ext_vector_type(16))) __bf16 v16bf;
typedef __attribute__((ext_vector_type(8)))  __bf16 v8bf;
typedef __attribute__((ext_vector_type(4)))  __bf16 v4bf;
typedef __attribute__((ext_vector_type(8)))  float  v8f;
typedef __attribute__((ext_vector_type(4)))  unsigned int u32x4;
typedef __attribute__((ext_vector_type(8)))  int i32x8;
typedef __attribute__((ext_vector_type(4)))  int i32x4;

union BV { v16bf v; v8bf h[2]; };

__device__ __forceinline__ v8f wmma_bf16(v16bf a, v16bf b, v8f c) {
  // D = A(16x32 bf16) * B(32x16 bf16) + C(16x16 f32)
  return __builtin_amdgcn_wmma_f32_16x16x32_bf16(false, a, false, b, (short)0, c,
                                                 false, false);
}

__device__ __forceinline__ v8f v8f_zero() {
  v8f z = {0.f, 0.f, 0.f, 0.f, 0.f, 0.f, 0.f, 0.f};
  return z;
}

// --------------------------------------------------------------- TDM (2-D tile)
// Build a Tensor DMA Descriptor (CDNA5 ISA) for a 2-D
// tile of 2-byte elements and issue TENSOR_LOAD_TO_LDS. dims/strides in elements.
__device__ __forceinline__ void tdm_load_2d(unsigned lds_byte_addr,
                                            const void* gptr,
                                            unsigned dim0, unsigned dim1,
                                            unsigned tile0, unsigned tile1,
                                            unsigned long long stride0) {
  unsigned long long ga = (unsigned long long)(uintptr_t)gptr;
  u32x4 g0;
  g0.x = 1u;                                              // count=1 user descriptor
  g0.y = lds_byte_addr;                                   // lds_addr
  g0.z = (unsigned)ga;                                    // global_addr[31:0]
  g0.w = (unsigned)((ga >> 32) & 0x01FFFFFFull) | (2u << 30);  // addr[56:32]|type=2
  i32x8 g1;
  g1[0] = (int)(1u << 16);                                // data_size=1 -> 2B elems
  g1[1] = (int)((dim0 & 0xFFFFu) << 16);                  // tensor_dim0 @bit48
  g1[2] = (int)(((dim0 >> 16) & 0xFFFFu) | ((dim1 & 0xFFFFu) << 16));
  g1[3] = (int)(((dim1 >> 16) & 0xFFFFu) | ((tile0 & 0xFFFFu) << 16));
  g1[4] = (int)(tile1 & 0xFFFFu);                         // tile_dim1 (tile_dim2=0)
  g1[5] = (int)(unsigned)(stride0 & 0xFFFFFFFFull);       // dim0_stride @bit160
  g1[6] = (int)(unsigned)((stride0 >> 32) & 0xFFFFull);   // stride[47:32]; d1_stride=0
  g1[7] = 0;
  i32x4 gz = {0, 0, 0, 0};
#if __has_include(<hip/amd_detail/amd_gfx1250_TDM.h>)
  i32x8 gz8 = {0, 0, 0, 0, 0, 0, 0, 0};
  __builtin_amdgcn_tensor_load_to_lds(g0, g1, gz, gz, gz8, 0);
#else
  __builtin_amdgcn_tensor_load_to_lds(g0, g1, gz, gz, 0);
#endif
}

__device__ __forceinline__ unsigned lds_addr_of(const void* p) {
  // flat shared pointer: low 32 bits are the LDS offset (ISA §10.2 aperture rule)
  return (unsigned)(uintptr_t)p;
}

// ---------------------------------------------------------------- cast x -> bf16
__global__ void cast_x_kernel(const float* __restrict__ x, __bf16* __restrict__ xb) {
  size_t i = ((size_t)blockIdx.x * blockDim.x + threadIdx.x) * 4;
  float4 f = *(const float4*)(x + i);
  v4bf o;
  o.x = (__bf16)f.x; o.y = (__bf16)f.y; o.z = (__bf16)f.z; o.w = (__bf16)f.w;
  *(v4bf*)(xb + i) = o;
}

// ------------------------------------------- weight transpose + cast: Wt[n][k]=W[k][n]
__global__ void transpose_cast_kernel(const float* __restrict__ w, __bf16* __restrict__ wt) {
  __shared__ float tile[32][33];
  const int tx = threadIdx.x, ty = threadIdx.y;
  const int n  = blockIdx.x * 32 + tx;
  const int k0 = blockIdx.y * 32;
#pragma unroll
  for (int j = 0; j < 32; j += 8)
    tile[ty + j][tx] = w[(size_t)(k0 + ty + j) * D_MODEL + n];
  __syncthreads();
  const int k  = k0 + tx;
  const int nO = blockIdx.x * 32 + ty;
#pragma unroll
  for (int j = 0; j < 32; j += 8)
    wt[(size_t)(nO + j) * D_MODEL + k] = (__bf16)tile[tx][ty + j];
}

// ------------------------------------------------------------------- QKV GEMM
// One wave computes a 32x64 tile (8 WMMA / k-step, B reused across 2 M-subtiles).
// blockIdx.z selects Q/K/V. Q,K written [B,H,S,Hd]; V written transposed [B,H,Hd,S].
__global__ void qkv_gemm_kernel(const __bf16* __restrict__ xb,
                                const __bf16* __restrict__ wtq,
                                const __bf16* __restrict__ wtk,
                                const __bf16* __restrict__ wtv,
                                __bf16* __restrict__ Qn,
                                __bf16* __restrict__ Kn,
                                __bf16* __restrict__ Vt) {
  const int wave = threadIdx.x >> 5, lane = threadIdx.x & 31;
  const int hf = lane >> 4, c = lane & 15;
  const int m0 = (blockIdx.x * 8 + wave) * 32;
  const int n0 = blockIdx.y * 64;
  const int which = blockIdx.z;
  const __bf16* wt = (which == 0) ? wtq : (which == 1 ? wtk : wtv);

  v8f acc[2][4];
#pragma unroll
  for (int mt = 0; mt < 2; ++mt)
#pragma unroll
    for (int nt = 0; nt < 4; ++nt) acc[mt][nt] = v8f_zero();

  for (int kb = 0; kb < D_MODEL; kb += 32) {
    BV av[2];
#pragma unroll
    for (int mt = 0; mt < 2; ++mt) {
      const __bf16* arow = xb + (size_t)(m0 + mt * 16 + c) * D_MODEL + kb;
      av[mt].h[0] = *(const v8bf*)(arow + hf * 8);
      av[mt].h[1] = *(const v8bf*)(arow + 16 + hf * 8);
    }
#pragma unroll
    for (int nt = 0; nt < 4; ++nt) {
      BV bv;
      const __bf16* brow = wt + (size_t)(n0 + nt * 16 + c) * D_MODEL + kb;
      bv.h[0] = *(const v8bf*)(brow + hf * 8);
      bv.h[1] = *(const v8bf*)(brow + 16 + hf * 8);
#pragma unroll
      for (int mt = 0; mt < 2; ++mt)
        acc[mt][nt] = wmma_bf16(av[mt].v, bv.v, acc[mt][nt]);
    }
  }

#pragma unroll
  for (int mt = 0; mt < 2; ++mt)
#pragma unroll
    for (int nt = 0; nt < 4; ++nt)
#pragma unroll
      for (int i = 0; i < 8; ++i) {
        const int m = m0 + mt * 16 + i + 8 * hf;  // token row
        const int j = n0 + nt * 16 + c;           // model column
        const int b = m >> 11, s = m & (SEQ - 1);
        const int hh = j >> 6, d = j & (HD - 1);
        const __bf16 val = (__bf16)acc[mt][nt][i];
        if (which == 2)
          Vt[(((size_t)b * HEADS + hh) * HD + d) * SEQ + s] = val;
        else if (which == 1)
          Kn[(((size_t)b * HEADS + hh) * SEQ + s) * HD + d] = val;
        else
          Qn[(((size_t)b * HEADS + hh) * SEQ + s) * HD + d] = val;
      }
}

// ------------------------------------------------------------- flash attention
// grid (S/64, B*H), block 128 (4 waves); each wave owns a 16-query tile.
// K/V 32-key blocks are TDM-staged into LDS (double-buffered, wave0 issues,
// s_wait_tensorcnt + block barrier publish) and shared by all 4 waves.
__global__ void attn_kernel(const __bf16* __restrict__ Qn,
                            const __bf16* __restrict__ Kn,
                            const __bf16* __restrict__ Vt,
                            __bf16* __restrict__ Hout) {
  __shared__ __attribute__((aligned(16))) __bf16 plds[4][16 * 32]; // per-wave P bounce
  __shared__ __attribute__((aligned(16))) __bf16 kbuf[2][32 * HD]; // [key][d]
  __shared__ __attribute__((aligned(16))) __bf16 vbuf[2][HD * 32]; // [d][key]
  const int wave = threadIdx.x >> 5, lane = threadIdx.x & 31;
  const int hf = lane >> 4, c = lane & 15;
  const int bh = blockIdx.y;
  const int q0 = (blockIdx.x * 4 + wave) * 16;
  const __bf16* Qb = Qn + (size_t)bh * SEQ * HD;
  const __bf16* Kb = Kn + (size_t)bh * SEQ * HD;
  const __bf16* Vb = Vt + (size_t)bh * HD * SEQ;
  __bf16* pl = plds[wave];

  // Q A-tiles for d-chunks [0,32) and [32,64): kept resident
  v16bf qa[2];
  {
    const __bf16* qrow = Qb + (size_t)(q0 + c) * HD;
#pragma unroll
    for (int kk = 0; kk < 2; ++kk) {
      BV t;
      t.h[0] = *(const v8bf*)(qrow + kk * 32 + hf * 8);
      t.h[1] = *(const v8bf*)(qrow + kk * 32 + 16 + hf * 8);
      qa[kk] = t.v;
    }
  }

  float mrow[8], ssum[8];
  v8f o[4];
#pragma unroll
  for (int i = 0; i < 8; ++i) { mrow[i] = -3.0e38f; ssum[i] = 0.f; }
#pragma unroll
  for (int t = 0; t < 4; ++t) o[t] = v8f_zero();

  // prologue: stage key-block 0
  if (wave == 0) {
    // K block: contiguous 32 rows x 64 d
    tdm_load_2d(lds_addr_of(kbuf[0]), Kb, HD, 32, HD, 32, HD);
    // V block: 64 d-rows x 32 keys, row stride SEQ (strided 2-D tile)
    tdm_load_2d(lds_addr_of(vbuf[0]), Vb, 32, HD, 32, HD, SEQ);
    __builtin_amdgcn_s_wait_tensorcnt(0);
  }
  __syncthreads();

  int cur = 0;
  for (int kb0 = 0; kb0 < SEQ; kb0 += 32) {
    const int nxt = cur ^ 1;
    if (wave == 0 && kb0 + 32 < SEQ) {  // prefetch next block while computing
      tdm_load_2d(lds_addr_of(kbuf[nxt]), Kb + (size_t)(kb0 + 32) * HD, HD, 32, HD, 32, HD);
      tdm_load_2d(lds_addr_of(vbuf[nxt]), Vb + (kb0 + 32), 32, HD, 32, HD, SEQ);
    }
    const __bf16* kB = kbuf[cur];
    const __bf16* vB = vbuf[cur];

    // ---- scores: 16 queries x 32 keys, K-dim = Hd = 64 (2 chunks)
    v8f sc[2] = {v8f_zero(), v8f_zero()};
#pragma unroll
    for (int t = 0; t < 2; ++t) {
#pragma unroll
      for (int kk = 0; kk < 2; ++kk) {
        BV bv;  // B(d x key): lane col = key t*16+c, d in table order from LDS K row
        const __bf16* krow = kB + (t * 16 + c) * HD + kk * 32;
        bv.h[0] = *(const v8bf*)(krow + hf * 8);
        bv.h[1] = *(const v8bf*)(krow + 16 + hf * 8);
        sc[t] = wmma_bf16(qa[kk], bv.v, sc[t]);
      }
    }
    // ---- online softmax update (rows live in 16-lane halves)
    float mx[8];
#pragma unroll
    for (int i = 0; i < 8; ++i) {
      sc[0][i] *= 0.125f;  // 1/sqrt(64)
      sc[1][i] *= 0.125f;
      mx[i] = fmaxf(sc[0][i], sc[1][i]);
    }
#pragma unroll
    for (int msk = 1; msk < 16; msk <<= 1)
#pragma unroll
      for (int i = 0; i < 8; ++i)
        mx[i] = fmaxf(mx[i], __shfl_xor(mx[i], msk, 32));
    float fac[8];
#pragma unroll
    for (int i = 0; i < 8; ++i) {
      const float mnew = fmaxf(mrow[i], mx[i]);
      fac[i] = __expf(mrow[i] - mnew);
      mrow[i] = mnew;
      ssum[i] *= fac[i];
    }
#pragma unroll
    for (int nt = 0; nt < 4; ++nt)
#pragma unroll
      for (int i = 0; i < 8; ++i) o[nt][i] *= fac[i];
    // ---- P = exp(s - m): C-layout -> row-major LDS (per-wave private 1KB)
#pragma unroll
    for (int t = 0; t < 2; ++t)
#pragma unroll
      for (int i = 0; i < 8; ++i) {
        const float p = __expf(sc[t][i] - mrow[i]);
        ssum[i] += p;  // per-lane partial row sum; reduced at the end
        pl[(i + 8 * hf) * 32 + t * 16 + c] = (__bf16)p;
      }
    asm volatile("s_wait_dscnt 0x0" ::: "memory");  // wave-private LDS handoff
    BV pa;  // A-layout reload of P (16x32)
    const __bf16* prow = pl + c * 32;
    pa.h[0] = *(const v8bf*)(prow + hf * 8);
    pa.h[1] = *(const v8bf*)(prow + 16 + hf * 8);
    // ---- O += P @ V  (B(key x d) from LDS V tile: keys in table order per lane)
#pragma unroll
    for (int nt = 0; nt < 4; ++nt) {
      BV bv;
      const __bf16* vrow = vB + (nt * 16 + c) * 32;
      bv.h[0] = *(const v8bf*)(vrow + hf * 8);
      bv.h[1] = *(const v8bf*)(vrow + 16 + hf * 8);
      o[nt] = wmma_bf16(pa.v, bv.v, o[nt]);
    }

    if (wave == 0) __builtin_amdgcn_s_wait_tensorcnt(0);  // DMA for nxt landed
    __syncthreads();                                      // publish; protect cur reuse
    cur = nxt;
  }

  // full row sums, normalize, store head_out [B,S,D] bf16
#pragma unroll
  for (int msk = 1; msk < 16; msk <<= 1)
#pragma unroll
    for (int i = 0; i < 8; ++i)
      ssum[i] += __shfl_xor(ssum[i], msk, 32);
  const int b = bh >> 4, hh = bh & 15;
#pragma unroll
  for (int nt = 0; nt < 4; ++nt)
#pragma unroll
    for (int i = 0; i < 8; ++i) {
      const int q = q0 + i + 8 * hf;
      const int d = nt * 16 + c;
      const float val = o[nt][i] / ssum[i];
      Hout[((size_t)b * SEQ + q) * D_MODEL + hh * HD + d] = (__bf16)val;
    }
}

// -------------------------------------------------------- final projection GEMM
__global__ void out_gemm_kernel(const __bf16* __restrict__ hb,
                                const __bf16* __restrict__ wtf,
                                float* __restrict__ out) {
  const int wave = threadIdx.x >> 5, lane = threadIdx.x & 31;
  const int hf = lane >> 4, c = lane & 15;
  const int m0 = (blockIdx.x * 8 + wave) * 32;
  const int n0 = blockIdx.y * 64;
  v8f acc[2][4];
#pragma unroll
  for (int mt = 0; mt < 2; ++mt)
#pragma unroll
    for (int nt = 0; nt < 4; ++nt) acc[mt][nt] = v8f_zero();

  for (int kb = 0; kb < D_MODEL; kb += 32) {
    BV av[2];
#pragma unroll
    for (int mt = 0; mt < 2; ++mt) {
      const __bf16* arow = hb + (size_t)(m0 + mt * 16 + c) * D_MODEL + kb;
      av[mt].h[0] = *(const v8bf*)(arow + hf * 8);
      av[mt].h[1] = *(const v8bf*)(arow + 16 + hf * 8);
    }
#pragma unroll
    for (int nt = 0; nt < 4; ++nt) {
      BV bv;
      const __bf16* brow = wtf + (size_t)(n0 + nt * 16 + c) * D_MODEL + kb;
      bv.h[0] = *(const v8bf*)(brow + hf * 8);
      bv.h[1] = *(const v8bf*)(brow + 16 + hf * 8);
#pragma unroll
      for (int mt = 0; mt < 2; ++mt)
        acc[mt][nt] = wmma_bf16(av[mt].v, bv.v, acc[mt][nt]);
    }
  }
#pragma unroll
  for (int mt = 0; mt < 2; ++mt)
#pragma unroll
    for (int nt = 0; nt < 4; ++nt)
#pragma unroll
      for (int i = 0; i < 8; ++i)
        out[(size_t)(m0 + mt * 16 + i + 8 * hf) * D_MODEL + n0 + nt * 16 + c] =
            acc[mt][nt][i];
}

// ------------------------------------------------------------------- launcher
extern "C" void kernel_launch(void* const* d_in, const int* in_sizes, int n_in,
                              void* d_out, int out_size, void* d_ws, size_t ws_size,
                              hipStream_t stream) {
  const float* x  = (const float*)d_in[0];
  const float* Wq = (const float*)d_in[1];
  const float* Wk = (const float*)d_in[2];
  const float* Wv = (const float*)d_in[3];
  const float* Wo = (const float*)d_in[4];

  char* ws = (char*)d_ws;
  size_t off = 0;
  const size_t XB_BYTES = (size_t)MTOT * D_MODEL * sizeof(__bf16);     // 8 MB
  const size_t W_BYTES  = (size_t)D_MODEL * D_MODEL * sizeof(__bf16);  // 2 MB
  __bf16* xb   = (__bf16*)(ws + off); off += XB_BYTES;
  __bf16* wtq  = (__bf16*)(ws + off); off += W_BYTES;
  __bf16* wtk  = (__bf16*)(ws + off); off += W_BYTES;
  __bf16* wtv  = (__bf16*)(ws + off); off += W_BYTES;
  __bf16* wtf  = (__bf16*)(ws + off); off += W_BYTES;
  __bf16* Qn   = (__bf16*)(ws + off); off += XB_BYTES;  // [B,H,S,Hd]
  __bf16* Kn   = (__bf16*)(ws + off); off += XB_BYTES;  // [B,H,S,Hd]
  __bf16* Vt   = (__bf16*)(ws + off); off += XB_BYTES;  // [B,H,Hd,S]
  __bf16* Hout = (__bf16*)(ws + off); off += XB_BYTES;  // [B,S,D]

  cast_x_kernel<<<(MTOT * D_MODEL / 4) / 256, 256, 0, stream>>>(x, xb);

  dim3 tb(32, 8), tg(D_MODEL / 32, D_MODEL / 32);
  transpose_cast_kernel<<<tg, tb, 0, stream>>>(Wq, wtq);
  transpose_cast_kernel<<<tg, tb, 0, stream>>>(Wk, wtk);
  transpose_cast_kernel<<<tg, tb, 0, stream>>>(Wv, wtv);
  transpose_cast_kernel<<<tg, tb, 0, stream>>>(Wo, wtf);

  qkv_gemm_kernel<<<dim3(MTOT / 256, D_MODEL / 64, 3), 256, 0, stream>>>(
      xb, wtq, wtk, wtv, Qn, Kn, Vt);

  attn_kernel<<<dim3(SEQ / 64, BATCH * HEADS), 128, 0, stream>>>(Qn, Kn, Vt, Hout);

  out_gemm_kernel<<<dim3(MTOT / 256, D_MODEL / 64), 256, 0, stream>>>(
      Hout, wtf, (float*)d_out);
}
